// Experts_84868553769176
// MI455X (gfx1250) — matrix operand order repaired and checked by reading, not compiled, not run on hardware
//
#include <hip/hip_runtime.h>

#define E_   64
#define I_   512
#define O_   512
#define T_   131072
#define CAP_ 3072

#define BM   128
#define BN   128
#define KC   64            // K-chunk staged in LDS per iteration
#define PAD  8
#define STR  (KC + PAD)    // 72 bf16 = 144 B row stride (16B aligned)
#define NTM  (CAP_ / BM)   // 24 row tiles per expert

typedef __attribute__((ext_vector_type(16))) __bf16 v16bf;
typedef __attribute__((ext_vector_type(8)))  __bf16 v8bf;
typedef __attribute__((ext_vector_type(4)))  __bf16 v4bf;
typedef __attribute__((ext_vector_type(2)))  __bf16 v2bf;
typedef __attribute__((ext_vector_type(8)))  float  v8f;
typedef __attribute__((ext_vector_type(4)))  float  v4f;

// f32x4 -> bf16x4 using the native packed convert when available.
static __device__ __forceinline__ v4bf cvt4(v4f v) {
#if __has_builtin(__builtin_amdgcn_cvt_pk_bf16_f32)
    v2bf lo = __builtin_amdgcn_cvt_pk_bf16_f32(v[0], v[1]);
    v2bf hi = __builtin_amdgcn_cvt_pk_bf16_f32(v[2], v[3]);
    return __builtin_shufflevector(lo, hi, 0, 1, 2, 3);
#else
    v4bf r = { (__bf16)v[0], (__bf16)v[1], (__bf16)v[2], (__bf16)v[3] };
    return r;
#endif
}

__global__ void moe_zero_counts(int* __restrict__ counts) {
    if (threadIdx.x < E_) counts[threadIdx.x] = 0;
}

__global__ void moe_route(const int* __restrict__ eid,
                          int* __restrict__ counts,
                          int* __restrict__ toklist,
                          float* __restrict__ out) {
    int t = blockIdx.x * blockDim.x + threadIdx.x;
    if (t >= T_) return;
    int e = eid[t];
    int pos = atomicAdd(&counts[e], 1);
    if (pos < CAP_) {
        toklist[e * CAP_ + pos] = t;
    } else {
        // capacity-dropped token -> zero output row (matches reference)
        float* o = out + (size_t)t * O_;
        for (int i = 0; i < O_; ++i) o[i] = 0.0f;
    }
}

__global__ void __launch_bounds__(256)
moe_gemm(const float* __restrict__ x,
         const float* __restrict__ w,
         const float* __restrict__ bias,
         const int*   __restrict__ counts,
         const int*   __restrict__ toklist,
         float*       __restrict__ out) {
    __shared__ __bf16 sX[BM * STR];
    __shared__ __bf16 sW[BN * STR];
    __shared__ int    sTok[BM];

    const int e  = blockIdx.x / NTM;
    const int tm = blockIdx.x % NTM;
    const int cnt = counts[e];
    const int rows_total = cnt < CAP_ ? cnt : CAP_;
    const int row0 = tm * BM;
    if (row0 >= rows_total) return;
    const int n0  = blockIdx.y * BN;
    const int tid = threadIdx.x;

    if (tid < BM) {
        int r = row0 + tid;
        sTok[tid] = (r < rows_total) ? toklist[e * CAP_ + r] : 0; // clamp: safe read
    }
    __syncthreads();

    // Per-thread global->LDS mapping: 2048 float4 per matrix per chunk, 8 each.
    int          off[8];
    const float* px[8];
    const float* pw[8];
#pragma unroll
    for (int i = 0; i < 8; ++i) {
        int id  = tid + i * 256;
        int row = id >> 4;            // 0..127
        int c4  = (id & 15) << 2;     // 0,4,...,60
        off[i] = row * STR + c4;
        px[i]  = x + (size_t)sTok[row] * I_ + c4;
        pw[i]  = w + ((size_t)e * O_ + (size_t)(n0 + row)) * I_ + c4;
    }

    const int wave  = tid >> 5;
    const int lane  = tid & 31;
    const int lhalf = lane >> 4;      // 0|1
    const int l15   = lane & 15;
    const int wm    = wave & 1;       // 2 row groups of 64
    const int wn    = wave >> 1;      // 4 col groups of 32

    v8f acc[4][2];
#pragma unroll
    for (int mt = 0; mt < 4; ++mt)
#pragma unroll
        for (int nt = 0; nt < 2; ++nt)
            acc[mt][nt] = (v8f)(0.0f);

    for (int k0 = 0; k0 < I_; k0 += KC) {
        // ---- stage chunk to LDS with fp32->bf16 conversion ----
#pragma unroll
        for (int i = 0; i < 8; ++i) {
            v4f vx = *(const v4f*)(px[i] + k0);
            v4f vw = *(const v4f*)(pw[i] + k0);
            *(v4bf*)&sX[off[i]] = cvt4(vx);
            *(v4bf*)&sW[off[i]] = cvt4(vw);
            if (k0 + KC < I_) {
                __builtin_prefetch(px[i] + k0 + KC, 0, 0);
                __builtin_prefetch(pw[i] + k0 + KC, 0, 0);
            }
        }
        __syncthreads();

        // ---- WMMA over the chunk, K steps of 32 ----
#pragma unroll
        for (int ks = 0; ks < KC; ks += 32) {
            // A fragments (16x32 bf16): lane<16 holds K {0..7,16..23}, lane>=16 {8..15,24..31}
            const int kA = ks + (lhalf ? 8 : 0);
            v16bf a[4];
#pragma unroll
            for (int mt = 0; mt < 4; ++mt) {
                int rowm = wm * 64 + mt * 16 + l15;
                v8bf lo = *(const v8bf*)&sX[rowm * STR + kA];
                v8bf hi = *(const v8bf*)&sX[rowm * STR + kA + 16];
                a[mt] = __builtin_shufflevector(lo, hi, 0, 1, 2, 3, 4, 5, 6, 7,
                                                        8, 9, 10, 11, 12, 13, 14, 15);
            }
            // B fragments (32x16 bf16): lane<16 K {0..7,16..23}, lane>=16 {8..15,24..31}; col = lane&15
            const int kB = ks + (lhalf ? 8 : 0);
            v16bf b[2];
#pragma unroll
            for (int nt = 0; nt < 2; ++nt) {
                int coln = wn * 32 + nt * 16 + l15;
                v8bf lo = *(const v8bf*)&sW[coln * STR + kB];
                v8bf hi = *(const v8bf*)&sW[coln * STR + kB + 16];
                b[nt] = __builtin_shufflevector(lo, hi, 0, 1, 2, 3, 4, 5, 6, 7,
                                                        8, 9, 10, 11, 12, 13, 14, 15);
            }
#pragma unroll
            for (int mt = 0; mt < 4; ++mt)
#pragma unroll
                for (int nt = 0; nt < 2; ++nt)
                    acc[mt][nt] = __builtin_amdgcn_wmma_f32_16x16x32_bf16(
                        false, a[mt], false, b[nt], (short)0, acc[mt][nt],
                        false, false);
        }
        __syncthreads();
    }

    // ---- epilogue: bias + scatter to out[token] ----
    float bv[2];
#pragma unroll
    for (int nt = 0; nt < 2; ++nt)
        bv[nt] = bias[e * O_ + n0 + wn * 32 + nt * 16 + l15];

#pragma unroll
    for (int mt = 0; mt < 4; ++mt) {
#pragma unroll
        for (int i = 0; i < 8; ++i) {
            int lm = wm * 64 + mt * 16 + (lhalf ? 8 + i : i);
            if (row0 + lm < rows_total) {
                int tok = sTok[lm];
                float* orow = out + (size_t)tok * O_;
#pragma unroll
                for (int nt = 0; nt < 2; ++nt) {
                    int col = n0 + wn * 32 + nt * 16 + l15;
                    orow[col] = acc[mt][nt][i] + bv[nt];
                }
            }
        }
    }
}

extern "C" void kernel_launch(void* const* d_in, const int* in_sizes, int n_in,
                              void* d_out, int out_size, void* d_ws, size_t ws_size,
                              hipStream_t stream) {
    const float* x    = (const float*)d_in[0];
    const float* w    = (const float*)d_in[1];
    const float* bias = (const float*)d_in[2];
    const int*   eid  = (const int*)d_in[3];
    float*       out  = (float*)d_out;

    int* counts  = (int*)d_ws;
    int* toklist = counts + 64;   // E*CAP ints = 768 KB of scratch

    moe_zero_counts<<<1, 64, 0, stream>>>(counts);
    moe_route<<<T_ / 256, 256, 0, stream>>>(eid, counts, toklist, out);

    dim3 grid(E_ * NTM, O_ / BN);
    moe_gemm<<<grid, 256, 0, stream>>>(x, w, bias, counts, toklist, out);
}
